// GoFeature_37787122270732
// MI455X (gfx1250) — matrix operand order repaired and checked by reading, not compiled, not run on hardware
//
#include <hip/hip_runtime.h>
#include <hip/hip_bf16.h>

// ---------- types ----------
typedef __bf16 bf16;
typedef __attribute__((ext_vector_type(16))) __bf16 bf16x16;
typedef __attribute__((ext_vector_type(8)))  __bf16 bf16x8;
typedef __attribute__((ext_vector_type(8)))  float  f32x8;
typedef __attribute__((ext_vector_type(4)))  float  f32x4;

#define N_ROWS 4096
#define IN_F   1024
#define CLS    512
#define OC     32

__device__ __forceinline__ bf16x16 combine8(bf16x8 lo, bf16x8 hi) {
  bf16x16 r;
#pragma unroll
  for (int e = 0; e < 8; ++e) { r[e] = lo[e]; r[e + 8] = hi[e]; }
  return r;
}

// A-fragment: one contiguous 16-element run (K = 16*hi .. +16)
__device__ __forceinline__ bf16x16 load_frag_a(const bf16* p) {
  return combine8(*(const bf16x8*)p, *(const bf16x8*)(p + 8));
}
// B-fragment: one contiguous 16-element run (K = 16*hi .. +16)
__device__ __forceinline__ bf16x16 load_frag_b(const bf16* p) {
  return combine8(*(const bf16x8*)p, *(const bf16x8*)(p + 8));
}

__device__ __forceinline__ f32x8 wmma_bf16(bf16x16 a, bf16x16 b, f32x8 c) {
  // D(16x16,f32) = A(16x32,bf16) * B(32x16,bf16) + C -> v_wmma_f32_16x16x32_bf16
  return __builtin_amdgcn_wmma_f32_16x16x32_bf16(false, a, false, b, (short)0, c,
                                                 false, false);
}

// ---------- kernel 0: fp32 -> bf16 conversion of x and W_w ----------
__global__ void cvt_bf16_kernel(const float* __restrict__ x,
                                const float* __restrict__ W,
                                bf16* __restrict__ xb, bf16* __restrict__ Wb) {
  const int nx = N_ROWS * IN_F;
  const int nw = CLS * IN_F;
  int stride = gridDim.x * blockDim.x;
  for (int i = blockIdx.x * blockDim.x + threadIdx.x; i < nx + nw; i += stride) {
    if (i < nx) xb[i] = (bf16)x[i];
    else        Wb[i - nx] = (bf16)W[i - nx];
  }
}

// ---------- kernel 1: h = x @ W_w^T + b  (writes h_bf16 and h^T_bf16) ----------
// Ping-pong double buffering (unroll-by-2): no backedge register copies,
// each WMMA group overlaps the other buffer's 10 in-flight loads.
__global__ __launch_bounds__(128) void linear_kernel(
    const bf16* __restrict__ xb, const bf16* __restrict__ Wb,
    const float* __restrict__ bias, bf16* __restrict__ hb,
    bf16* __restrict__ htb) {
  const int tid = threadIdx.x;
  const int wave = tid >> 5, lane = tid & 31, l15 = lane & 15, hi = lane >> 4;
  const int i0 = blockIdx.x * 16;                 // 16 output rows
  const int c0 = blockIdx.y * 256 + wave * 64;    // 64 output cols per wave

  f32x8 zero = {0, 0, 0, 0, 0, 0, 0, 0};
  f32x8 acc[4];
#pragma unroll
  for (int ct = 0; ct < 4; ++ct) acc[ct] = zero;

  const bf16* arow = xb + (i0 + l15) * IN_F + 16 * hi;
  const bf16* brow[4];
#pragma unroll
  for (int ct = 0; ct < 4; ++ct)
    brow[ct] = Wb + (c0 + ct * 16 + l15) * IN_F + 16 * hi;

  // prologue: buffer set 0 holds kk = 0
  bf16x16 a0 = load_frag_a(arow);
  bf16x16 b0[4];
#pragma unroll
  for (int ct = 0; ct < 4; ++ct) b0[ct] = load_frag_b(brow[ct]);

  for (int kk = 0; kk < IN_F; kk += 64) {
    // prefetch set 1 @ kk+32 (always valid: IN_F is a multiple of 64)
    bf16x16 a1 = load_frag_a(arow + kk + 32);
    bf16x16 b1[4];
#pragma unroll
    for (int ct = 0; ct < 4; ++ct) b1[ct] = load_frag_b(brow[ct] + kk + 32);
    // consume set 0 while set 1 is in flight
#pragma unroll
    for (int ct = 0; ct < 4; ++ct) acc[ct] = wmma_bf16(a0, b0[ct], acc[ct]);
    // prefetch set 0 @ kk+64 (dummy wrap on the final step)
    const int kn = (kk + 64 < IN_F) ? kk + 64 : 0;
    a0 = load_frag_a(arow + kn);
#pragma unroll
    for (int ct = 0; ct < 4; ++ct) b0[ct] = load_frag_b(brow[ct] + kn);
    // consume set 1 while set 0 is in flight
#pragma unroll
    for (int ct = 0; ct < 4; ++ct) acc[ct] = wmma_bf16(a1, b1[ct], acc[ct]);
  }

#pragma unroll
  for (int ct = 0; ct < 4; ++ct) {
    int col = c0 + ct * 16 + l15;
    float bv = bias[col];
#pragma unroll
    for (int v = 0; v < 8; ++v) {
      int row = i0 + v + 8 * hi;
      bf16 hv = (bf16)(acc[ct][v] + bv);
      hb[row * CLS + col] = hv;        // h    [4096][512]
      htb[col * N_ROWS + row] = hv;    // h^T  [512][4096]
    }
  }
}

// ---------- kernel 2: fused  a = relu(tanh(softmax(tanh(h h^T)) @ h)) ----------
// tanh bounds scores to [-1,1] -> softmax needs no running max: single pass.
__global__ __launch_bounds__(128) void attn_kernel(
    const bf16* __restrict__ hb, const bf16* __restrict__ htb,
    float* __restrict__ a_out) {
  __shared__ bf16  lds_hi[16 * 512];          // h rows i0..i0+15 (bf16)
  __shared__ float lds_sp[4][2][16][16];      // per-wave score partials
  __shared__ bf16  lds_E[16 * 32];            // exp(tanh(s)) tile, bf16
  __shared__ float lds_dsum[128];
  __shared__ float lds_denom[16];

  const int tid = threadIdx.x;
  const int wave = tid >> 5, lane = tid & 31, l15 = lane & 15, hi = lane >> 4;
  const int i0 = blockIdx.x * 16;
  const int c0 = wave * 128;                  // 128 output cols per wave

  {   // stage the 16x512 bf16 h_i tile (16 KB, contiguous)
    const bf16x8* src = (const bf16x8*)(hb + i0 * CLS);
    bf16x8* dst = (bf16x8*)lds_hi;
    for (int t = tid; t < 1024; t += 128) dst[t] = src[t];
  }

  f32x8 zero = {0, 0, 0, 0, 0, 0, 0, 0};
  f32x8 nacc[8];
#pragma unroll
  for (int ct = 0; ct < 8; ++ct) nacc[ct] = zero;
  float dsum = 0.f;
  __syncthreads();

  // A-fragments of the scores GEMM are j-invariant: hoist out of the loop.
  bf16x16 afv[4];
#pragma unroll
  for (int c = 0; c < 4; ++c)
    afv[c] = load_frag_a(lds_hi + l15 * 512 + wave * 128 + c * 32 + 16 * hi);

  for (int j0 = 0; j0 < N_ROWS; j0 += 32) {
    // ---- phase 1: partial scores over this wave's K range (128 of 512) ----
    // batch all 16 global b128 loads, then run 8 WMMAs back-to-back
    bf16x16 bf0[4], bf1[4];
#pragma unroll
    for (int c = 0; c < 4; ++c) {
      int kb = wave * 128 + c * 32;
      bf0[c] = load_frag_b(hb + (j0 + l15) * CLS + kb + 16 * hi);
      bf1[c] = load_frag_b(hb + (j0 + 16 + l15) * CLS + kb + 16 * hi);
    }
    f32x8 s0 = zero, s1 = zero;
#pragma unroll
    for (int c = 0; c < 4; ++c) {
      s0 = wmma_bf16(afv[c], bf0[c], s0);
      s1 = wmma_bf16(afv[c], bf1[c], s1);
    }
#pragma unroll
    for (int v = 0; v < 8; ++v) {
      lds_sp[wave][0][v + 8 * hi][l15] = s0[v];
      lds_sp[wave][1][v + 8 * hi][l15] = s1[v];
    }
    __syncthreads();

    // ---- phase 2: cross-wave reduce, E = exp(tanh(s)), denom accumulation ----
    {
      int m = tid >> 3;            // row 0..15
      int jj = (tid & 7) * 4;      // col 0..28 step 4
      int t = jj >> 4, cc = jj & 15;
      f32x4 s = {0, 0, 0, 0};
#pragma unroll
      for (int w4 = 0; w4 < 4; ++w4) s += *(const f32x4*)&lds_sp[w4][t][m][cc];
#pragma unroll
      for (int q = 0; q < 4; ++q) {
        float ev = __expf(tanhf(s[q]));
        bf16 eb = (bf16)ev;
        dsum += (float)eb;               // consistent with bf16 numerator
        lds_E[m * 32 + jj + q] = eb;
      }
    }
    __syncthreads();

    // ---- phase 3: numer += E(16x32) @ h_j(32x512), this wave's 128 cols ----
    bf16x16 ef = load_frag_a(lds_E + l15 * 32 + 16 * hi);
    bf16x16 bfrag[8];
#pragma unroll
    for (int ct = 0; ct < 8; ++ct)
      bfrag[ct] = load_frag_b(htb + (c0 + ct * 16 + l15) * N_ROWS + j0 + 16 * hi);
#pragma unroll
    for (int ct = 0; ct < 8; ++ct)
      nacc[ct] = wmma_bf16(ef, bfrag[ct], nacc[ct]);
  }

  // ---- denominators: per-lane sums -> per-row ----
  lds_dsum[tid] = dsum;
  __syncthreads();
  if (tid < 16) {
    float s = 0.f;
#pragma unroll
    for (int t = 0; t < 8; ++t) s += lds_dsum[tid * 8 + t];
    lds_denom[tid] = s;
  }
  __syncthreads();

#pragma unroll
  for (int ct = 0; ct < 8; ++ct) {
    int col = c0 + ct * 16 + l15;
#pragma unroll
    for (int v = 0; v < 8; ++v) {
      int m = v + 8 * hi;
      float val = tanhf(nacc[ct][v] / lds_denom[m]);
      a_out[(i0 + m) * CLS + col] = fmaxf(val, 0.f);
    }
  }
}

// ---------- kernel 3: fused conv1(1->32,k3,p1)+relu, conv2(32->32,k3,p1)+relu,
//            output transposed to [n][p][o]; conv1 intermediate stays in LDS ----
__global__ __launch_bounds__(128) void conv_kernel(
    const float* __restrict__ a, const float* __restrict__ w1g,
    const float* __restrict__ b1g, const float* __restrict__ w2g,
    const float* __restrict__ b2g, float* __restrict__ out) {
  __shared__ float apad[514];
  __shared__ float z1[32][66];     // channel x (64 positions + halo)
  __shared__ float w1[96], b1s[32], w2t[3072], b2s[32];

  const int n = blockIdx.x, tid = threadIdx.x;
  for (int i = tid; i < 96; i += 128) w1[i] = w1g[i];
  for (int i = tid; i < 32; i += 128) { b1s[i] = b1g[i]; b2s[i] = b2g[i]; }
  for (int i = tid; i < 3072; i += 128) {   // transpose w2 -> [c][k][o]
    int o = i / 96, r = i % 96, c = r / 3, k = r % 3;
    w2t[(c * 3 + k) * 32 + o] = w2g[i];
  }
  if (tid == 0) { apad[0] = 0.f; apad[513] = 0.f; }
  for (int p = tid; p < 512; p += 128) apad[p + 1] = a[n * 512 + p];
  __syncthreads();

  for (int p0 = 0; p0 < 512; p0 += 64) {
    for (int idx = tid; idx < 32 * 66; idx += 128) {
      int c = idx / 66, qq = idx % 66, q = p0 - 1 + qq;
      float v = 0.f;
      if (q >= 0 && q < 512) {
        v = w1[3 * c] * apad[q] + w1[3 * c + 1] * apad[q + 1] +
            w1[3 * c + 2] * apad[q + 2] + b1s[c];
        v = fmaxf(v, 0.f);
      }
      z1[c][qq] = v;
    }
    __syncthreads();
    for (int idx = tid; idx < 64 * 32; idx += 128) {
      int lp = idx >> 5, o = idx & 31, p = p0 + lp;
      float s = b2s[o];
#pragma unroll
      for (int c = 0; c < 32; ++c) {
        s += w2t[(c * 3 + 0) * 32 + o] * z1[c][lp] +
             w2t[(c * 3 + 1) * 32 + o] * z1[c][lp + 1] +
             w2t[(c * 3 + 2) * 32 + o] * z1[c][lp + 2];
      }
      out[(n * 512 + p) * 32 + o] = fmaxf(s, 0.f);
    }
    __syncthreads();
  }
}

// ---------- launch ----------
extern "C" void kernel_launch(void* const* d_in, const int* in_sizes, int n_in,
                              void* d_out, int out_size, void* d_ws,
                              size_t ws_size, hipStream_t stream) {
  const float* x   = (const float*)d_in[0];
  const float* W_w = (const float*)d_in[1];
  const float* W_b = (const float*)d_in[2];
  const float* c1w = (const float*)d_in[3];
  const float* c1b = (const float*)d_in[4];
  const float* c2w = (const float*)d_in[5];
  const float* c2b = (const float*)d_in[6];
  float* out = (float*)d_out;

  char* ws = (char*)d_ws;
  bf16* xb   = (bf16*)(ws + 0);                       // 8 MB
  bf16* Wb   = (bf16*)(ws + 8388608);                 // 1 MB
  bf16* hb   = (bf16*)(ws + 9437184);                 // 4 MB
  bf16* htb  = (bf16*)(ws + 13631488);                // 4 MB
  float* a   = (float*)(ws + 17825792);               // 8 MB  (total 25 MB)

  cvt_bf16_kernel<<<2048, 256, 0, stream>>>(x, W_w, xb, Wb);
  linear_kernel<<<dim3(N_ROWS / 16, 2), 128, 0, stream>>>(xb, Wb, W_b, hb, htb);
  attn_kernel<<<N_ROWS / 16, 128, 0, stream>>>(hb, htb, a);
  conv_kernel<<<N_ROWS, 128, 0, stream>>>(a, c1w, c1b, c2w, c2b, out);
}
